// X3DAttention_67018669686802
// MI455X (gfx1250) — compile-verified, matrix operand order checked by reading, not run on hardware
//
#include <hip/hip_runtime.h>
#include <hip/hip_bf16.h>

typedef _Float16 h16;
typedef __attribute__((ext_vector_type(8)))  _Float16 v8h;
typedef __attribute__((ext_vector_type(16))) _Float16 v16h;
typedef __attribute__((ext_vector_type(8)))  float    v8f;

#define LL   384
#define CC   256
#define HH   8
#define DD   32
#define DD2  128
#define EPSV 1e-5f
#define SCL  0.17677669529663687f   // 1/sqrt(32)

// ---- WMMA fragment helpers ------------------------------------------------
// Row-major source (stride in halves). Follows CDNA5 16-bit A/B 16x32 layout:
// lane 0-15: v0..3 = K0..7,  v4..7 = K16..23 ; lane 16-31: K8..15 / K24..31.
__device__ __forceinline__ v16h frag_rowmajor(const h16* base, int stride,
                                              int row0, int k0, int lane) {
  int r = row0 + (lane & 15);
  const h16* p = base + (size_t)r * stride + k0 + ((lane & 16) ? 8 : 0);
  v8h lo = *(const v8h*)p;          // 16B: K+0..7 (or +8..15)
  v8h hi = *(const v8h*)(p + 16);   // 16B: K+16..23 (or +24..31)
  return __builtin_shufflevector(lo, hi, 0,1,2,3,4,5,6,7,8,9,10,11,12,13,14,15);
}

__device__ __forceinline__ v8f wmma_f16(v16h a, v16h b, v8f c) {
  return __builtin_amdgcn_wmma_f32_16x16x32_f16(false, a, false, b,
                                                (short)0, c, false, false);
}

// ---- weight transpose + f16 convert: W[K][N] -> WT[N][K] -------------------
__global__ void wt_convert(const float* __restrict__ W, h16* __restrict__ WT,
                           int K, int N) {
  int gid = blockIdx.x * 256 + threadIdx.x;
  if (gid < K * N) {
    int k = gid / N, n = gid - k * N;
    WT[(size_t)n * K + k] = (h16)W[gid];
  }
}

// ---- q,k,v projections -> [H][L][d] layout; also zero X2 accumulator ------
__global__ void qkv_kernel(const float* __restrict__ x3d,
                           const float* __restrict__ Wq, const float* __restrict__ bq,
                           const float* __restrict__ Wk, const float* __restrict__ bk,
                           const float* __restrict__ Wv, const float* __restrict__ bv,
                           float* __restrict__ Q, float* __restrict__ Kd,
                           float* __restrict__ V, float* __restrict__ X2) {
  int gid = blockIdx.x * 256 + threadIdx.x;       // 384*256
  int row = gid >> 8, oc = gid & 255;
  const float* x = x3d + (size_t)row * CC;
  float aq = bq[oc], ak = bk[oc], av = bv[oc];
  for (int c = 0; c < CC; ++c) {
    float xv = x[c];
    aq += xv * Wq[c * CC + oc];
    ak += xv * Wk[c * CC + oc];
    av += xv * Wv[c * CC + oc];
  }
  int h = oc >> 5, dd = oc & 31;
  int idx = (h * LL + row) * DD + dd;
  Q[idx] = aq; Kd[idx] = ak; V[idx] = av;
  X2[gid] = 0.f;
}

// ---- seed SC[h][i][j] with q_i . k_j (unscaled) ---------------------------
__global__ void qk_kernel(const float* __restrict__ Q, const float* __restrict__ Kd,
                          float* __restrict__ SC) {
  int gid = blockIdx.x * 256 + threadIdx.x;       // H*L*L
  int j = gid % LL; int hi = gid / LL; int i = hi % LL; int h = hi / LL;
  const float* q = Q  + (size_t)(h * LL + i) * DD;
  const float* k = Kd + (size_t)(h * LL + j) * DD;
  float s = 0.f;
  for (int dd = 0; dd < DD; ++dd) s += q[dd] * k[dd];
  SC[gid] = s;
}

// ---- fused pair kernel: the heavy WMMA workhorse --------------------------
// Block tile: 16 i-values x 4 j-values = 64 pair rows, full C=256 channels.
// Per WG: async-DMA x2d tile -> LDS, GEMM1 (64x128 @ 128x256), LN+ReLU,
// GEMM2/3 (64x256 @ 256x256), then scores (+= efk.q_j, scale, write SC)
// and x2 f32 global atomics.
__global__ __launch_bounds__(256, 1)
void pair_kernel(const float* __restrict__ x2d,
                 const h16*  __restrict__ WpreT, const float* __restrict__ bpre,
                 const float* __restrict__ ln_g, const float* __restrict__ ln_b,
                 const h16*  __restrict__ WeqT,  const float* __restrict__ beq,
                 const h16*  __restrict__ WekT,  const float* __restrict__ bek,
                 const float* __restrict__ Q,
                 float* __restrict__ SC, float* __restrict__ X2) {
  extern __shared__ char smem[];
  h16*   sA   = (h16*)  (smem);            // 64x128 f16      (16 KB)
  float* sE32 = (float*)(smem + 16384);    // 64x256 f32      (64 KB) -> reused as EFQ
  h16*   sE16 = (h16*)  (smem + 81920);    // 64x256 f16      (32 KB)
  float* sEFK = (float*)(smem + 114688);   // 64x256 f32      (64 KB)
  float* sQJ  = (float*)(smem + 180224);   // 4x256 f32       (4 KB)
  float* sS   = (float*)(smem + 184320);   // 64x8  f32       (2 KB)
  float* sRed = (float*)(smem + 186368);   // 64x4x2 f32      (2 KB)
  float* sEFQ = sE32;
  float* sStage = sEFK;                    // x2d f32 staging (32 KB, dead region)

  const int tid = threadIdx.x, lane = tid & 31, wave = tid >> 5;
  const int i0 = (blockIdx.x / 96) * 16;
  const int j0 = (blockIdx.x % 96) * 4;

  // Pull f16 weights for GEMM2/3 toward cache early (global_prefetch_b8).
  __builtin_prefetch(WeqT + (size_t)tid * 256, 0, 0);
  __builtin_prefetch(WekT + (size_t)tid * 256, 0, 0);

  // ---- stage x2d tile (f32) into LDS via gfx1250 async DMA ----------------
  // 64 rows x 128 f32 = 32 KB; 8 async b128 ops per wave, per-lane addresses.
  for (int n = 0; n < 8; ++n) {
    int t  = n * 256 + tid;          // flat 16B chunk index (0..2047)
    int p  = t >> 5;                 // pair row 0..63
    int kk = (t & 31) << 2;          // float offset within row
    int il = p >> 2, jl = p & 3;
    unsigned long long ga = (unsigned long long)(const void*)(x2d +
        (((size_t)(i0 + il)) * LL + (j0 + jl)) * DD2 + kk);
    unsigned ldsa = (unsigned)(size_t)(sStage + (t << 2));
    asm volatile("global_load_async_to_lds_b128 %0, %1, off"
                 :: "v"(ldsa), "v"(ga) : "memory");
  }
  // q rows for the 4 j's (regular loads)
  for (int t = tid; t < 4 * 256; t += 256) {
    int jl = t >> 8, ch = t & 255, h = ch >> 5, dd = ch & 31;
    sQJ[t] = Q[(size_t)(h * LL + (j0 + jl)) * DD + dd];
  }
  asm volatile("s_wait_asynccnt 0x0" ::: "memory");
  __syncthreads();

  // f32 -> f16 conversion for GEMM1 A operand
  for (int t = tid; t < 2048; t += 256) {
    const float4 v = *(const float4*)(sStage + (t << 2));
    h16* d = sA + (t << 2);
    d[0] = (h16)v.x; d[1] = (h16)v.y; d[2] = (h16)v.z; d[3] = (h16)v.w;
  }
  __syncthreads();

  const int mt  = wave & 3;          // M-tile (4 x 16 rows)
  const int nt0 = (wave >> 2) * 8;   // N-tile base (8 tiles per wave)

  // GEMM1: e = sA(64x128) @ WpreT^T + bpre -> sE32
  {
    v8f acc[8];
    v8f zero = {0.f,0.f,0.f,0.f,0.f,0.f,0.f,0.f};
    for (int t = 0; t < 8; ++t) acc[t] = zero;
    for (int kt = 0; kt < 4; ++kt) {
      v16h a = frag_rowmajor(sA, 128, mt * 16, kt * 32, lane);
      for (int t = 0; t < 8; ++t) {
        v16h b = frag_rowmajor(WpreT, 128, (nt0 + t) * 16, kt * 32, lane);
        acc[t] = wmma_f16(a, b, acc[t]);
      }
    }
    for (int t = 0; t < 8; ++t) {
      int col = (nt0 + t) * 16 + (lane & 15);
      float bv = bpre[col];
      int rb = mt * 16 + ((lane & 16) ? 8 : 0);
      for (int r = 0; r < 8; ++r) sE32[(rb + r) * 256 + col] = acc[t][r] + bv;
    }
  }
  __syncthreads();

  // LayerNorm + ReLU per pair row, write f16 -> sE16
  {
    int row = tid & 63, qtr = tid >> 6;
    const float* er = sE32 + row * 256 + qtr * 64;
    float s = 0.f, s2 = 0.f;
    for (int c = 0; c < 64; ++c) { float v = er[c]; s += v; s2 += v * v; }
    sRed[(row * 4 + qtr) * 2 + 0] = s;
    sRed[(row * 4 + qtr) * 2 + 1] = s2;
    __syncthreads();
    float ts = 0.f, ts2 = 0.f;
    for (int q2 = 0; q2 < 4; ++q2) {
      ts  += sRed[(row * 4 + q2) * 2 + 0];
      ts2 += sRed[(row * 4 + q2) * 2 + 1];
    }
    float mu   = ts * (1.f / 256.f);
    float var  = ts2 * (1.f / 256.f) - mu * mu;
    float rstd = rsqrtf(var + EPSV);
    for (int c = 0; c < 64; ++c) {
      int ch = qtr * 64 + c;
      float v = (sE32[row * 256 + ch] - mu) * rstd * ln_g[ch] + ln_b[ch];
      v = v > 0.f ? v : 0.f;
      sE16[row * 256 + ch] = (h16)v;
    }
  }
  __syncthreads();

  // GEMM2/3: efq = e @ Weq + beq ; efk = e @ Wek + bek
  for (int which = 0; which < 2; ++which) {
    const h16*   WT   = which ? WekT : WeqT;
    const float* bias = which ? bek  : beq;
    float*       out  = which ? sEFK : sEFQ;
    v8f acc[8];
    v8f zero = {0.f,0.f,0.f,0.f,0.f,0.f,0.f,0.f};
    for (int t = 0; t < 8; ++t) acc[t] = zero;
    for (int kt = 0; kt < 8; ++kt) {
      v16h a = frag_rowmajor(sE16, 256, mt * 16, kt * 32, lane);
      for (int t = 0; t < 8; ++t) {
        v16h b = frag_rowmajor(WT, 256, (nt0 + t) * 16, kt * 32, lane);
        acc[t] = wmma_f16(a, b, acc[t]);
      }
    }
    for (int t = 0; t < 8; ++t) {
      int col = (nt0 + t) * 16 + (lane & 15);
      float bv = bias[col];
      int rb = mt * 16 + ((lane & 16) ? 8 : 0);
      for (int r = 0; r < 8; ++r) out[(rb + r) * 256 + col] = acc[t][r] + bv;
    }
  }
  __syncthreads();

  // scores: s = (qk + sum_d efk[p,h,d]*q[h,j,d]) * SCALE ; write SC + sS
  for (int t = tid; t < 64 * 8; t += 256) {
    int p = t >> 3, h = t & 7;
    int il = p >> 2, jl = p & 3;
    const float* ek = sEFK + p * 256 + h * 32;
    const float* qj = sQJ + jl * 256 + h * 32;
    float s = 0.f;
    for (int dd = 0; dd < 32; ++dd) s += ek[dd] * qj[dd];
    size_t sci = ((size_t)h * LL + (i0 + il)) * LL + (j0 + jl);
    float sf = (SC[sci] + s) * SCL;
    SC[sci] = sf;
    sS[t] = sf;
  }
  __syncthreads();

  // x2[h,i,d] += sum_jl s[p,h] * efq[p, h*32+d]  (f32 global atomics)
  for (int t = tid; t < 16 * 256; t += 256) {
    int il = t >> 8, ch = t & 255, h = ch >> 5;
    float a = 0.f;
    for (int jl = 0; jl < 4; ++jl) {
      int p = il * 4 + jl;
      a += sS[p * 8 + h] * sEFQ[p * 256 + ch];
    }
    atomicAdd(&X2[(i0 + il) * 256 + ch], a);
  }
}

// ---- softmax over j, x1 = w@v, combine with x2 ----------------------------
__global__ __launch_bounds__(128)
void attn_kernel(const float* __restrict__ SC, const float* __restrict__ V,
                 const float* __restrict__ X2, float* __restrict__ XO) {
  __shared__ float sw[LL];
  __shared__ float sred[128];
  int h = blockIdx.x / LL, i = blockIdx.x % LL;
  int tid = threadIdx.x;
  const float* row = SC + ((size_t)h * LL + i) * LL;

  float m = -1e30f;
  for (int j = tid; j < LL; j += 128) m = fmaxf(m, row[j]);
  sred[tid] = m; __syncthreads();
  for (int s = 64; s > 0; s >>= 1) {
    if (tid < s) sred[tid] = fmaxf(sred[tid], sred[tid + s]);
    __syncthreads();
  }
  m = sred[0]; __syncthreads();

  float sum = 0.f;
  for (int j = tid; j < LL; j += 128) {
    float e = __expf(row[j] - m);
    sw[j] = e; sum += e;
  }
  sred[tid] = sum; __syncthreads();
  for (int s = 64; s > 0; s >>= 1) {
    if (tid < s) sred[tid] += sred[tid + s];
    __syncthreads();
  }
  float rs = 1.f / sred[0];
  __syncthreads();

  // x1[h,i,dd] = (sum_j w[j] * v[h,j,dd]) / Z
  int dd = tid & 31, part = tid >> 5;   // 4 partial sums per dd
  float acc = 0.f;
  const float* vh = V + (size_t)h * LL * DD;
  for (int j = part * 96; j < part * 96 + 96; ++j) acc += sw[j] * vh[j * DD + dd];
  sred[tid] = acc; __syncthreads();
  if (part == 0) {
    float x1 = (sred[dd] + sred[dd + 32] + sred[dd + 64] + sred[dd + 96]) * rs;
    int ch = h * 32 + dd;
    XO[i * CC + ch] = x1 + X2[i * CC + ch];
  }
}

// ---- output projection: OUT = XO @ Wo + bo (WMMA) -------------------------
__global__ __launch_bounds__(256, 1)
void outproj_kernel(const float* __restrict__ XO, const h16* __restrict__ WoT,
                    const float* __restrict__ bo, float* __restrict__ OUT) {
  __shared__ h16 sX[64 * 256];   // 32 KB
  int tid = threadIdx.x, lane = tid & 31, wave = tid >> 5;
  int row0 = blockIdx.x * 64;
  for (int t = tid; t < 64 * 256 / 4; t += 256) {
    int r = t >> 6, kk = (t & 63) << 2;
    float4 v = *(const float4*)(XO + (size_t)(row0 + r) * 256 + kk);
    h16* d = sX + r * 256 + kk;
    d[0] = (h16)v.x; d[1] = (h16)v.y; d[2] = (h16)v.z; d[3] = (h16)v.w;
  }
  __syncthreads();
  int mt = wave & 3, nt0 = (wave >> 2) * 8;
  v8f acc[8];
  v8f zero = {0.f,0.f,0.f,0.f,0.f,0.f,0.f,0.f};
  for (int t = 0; t < 8; ++t) acc[t] = zero;
  for (int kt = 0; kt < 8; ++kt) {
    v16h a = frag_rowmajor(sX, 256, mt * 16, kt * 32, lane);
    for (int t = 0; t < 8; ++t) {
      v16h b = frag_rowmajor(WoT, 256, (nt0 + t) * 16, kt * 32, lane);
      acc[t] = wmma_f16(a, b, acc[t]);
    }
  }
  for (int t = 0; t < 8; ++t) {
    int col = (nt0 + t) * 16 + (lane & 15);
    float bv = bo[col];
    int rb = mt * 16 + ((lane & 16) ? 8 : 0);
    for (int r = 0; r < 8; ++r)
      OUT[(size_t)(row0 + rb + r) * 256 + col] = acc[t][r] + bv;
  }
}

// ---------------------------------------------------------------------------
extern "C" void kernel_launch(void* const* d_in, const int* in_sizes, int n_in,
                              void* d_out, int out_size, void* d_ws, size_t ws_size,
                              hipStream_t stream) {
  (void)in_sizes; (void)n_in; (void)out_size; (void)ws_size;
  const float* x2d  = (const float*)d_in[0];
  const float* x3d  = (const float*)d_in[1];
  const float* Wq   = (const float*)d_in[2];
  const float* bq   = (const float*)d_in[3];
  const float* Wk   = (const float*)d_in[4];
  const float* bk   = (const float*)d_in[5];
  const float* Wv   = (const float*)d_in[6];
  const float* bv   = (const float*)d_in[7];
  const float* Wpre = (const float*)d_in[8];
  const float* bpre = (const float*)d_in[9];
  const float* ln_g = (const float*)d_in[10];
  const float* ln_b = (const float*)d_in[11];
  const float* Weq  = (const float*)d_in[12];
  const float* beq  = (const float*)d_in[13];
  const float* Wek  = (const float*)d_in[14];
  const float* bek  = (const float*)d_in[15];
  const float* Wo   = (const float*)d_in[16];
  const float* bo   = (const float*)d_in[17];

  char* ws = (char*)d_ws;
  float* Q     = (float*)(ws + 0);         // [H][L][d]
  float* Kd    = (float*)(ws + 393216);
  float* V     = (float*)(ws + 786432);
  float* SC    = (float*)(ws + 1179648);   // [H][L][L]
  float* X2    = (float*)(ws + 5898240);   // [L][C]
  float* XO    = (float*)(ws + 6291456);   // [L][C]
  h16*   WpreT = (h16*)  (ws + 6684672);   // [256][128]
  h16*   WeqT  = (h16*)  (ws + 6750208);   // [256][256]
  h16*   WekT  = (h16*)  (ws + 6881280);
  h16*   WoT   = (h16*)  (ws + 7012352);

  (void)hipFuncSetAttribute((const void*)pair_kernel,
                            hipFuncAttributeMaxDynamicSharedMemorySize, 188416);

  wt_convert<<<128, 256, 0, stream>>>(Wpre, WpreT, 128, 256);
  wt_convert<<<256, 256, 0, stream>>>(Weq,  WeqT,  256, 256);
  wt_convert<<<256, 256, 0, stream>>>(Wek,  WekT,  256, 256);
  wt_convert<<<256, 256, 0, stream>>>(Wo,   WoT,   256, 256);

  qkv_kernel<<<LL, 256, 0, stream>>>(x3d, Wq, bq, Wk, bk, Wv, bv, Q, Kd, V, X2);
  qk_kernel<<<HH * LL * LL / 256, 256, 0, stream>>>(Q, Kd, SC);

  pair_kernel<<<(LL / 16) * (LL / 4), 256, 188416, stream>>>(
      x2d, WpreT, bpre, ln_g, ln_b, WeqT, beq, WekT, bek, Q, SC, X2);

  attn_kernel<<<HH * LL, 128, 0, stream>>>(SC, V, X2, XO);
  outproj_kernel<<<LL / 64, 256, 0, stream>>>(XO, WoT, bo, (float*)d_out);
}
